// Attention_63934883168272
// MI455X (gfx1250) — matrix operand order repaired and checked by reading, not compiled, not run on hardware
//
#include <hip/hip_runtime.h>
#include <hip/hip_bf16.h>

typedef __attribute__((ext_vector_type(16))) __bf16 v16bf;
typedef __attribute__((ext_vector_type(8)))  __bf16 v8bf;
typedef __attribute__((ext_vector_type(8)))  float  v8f;
typedef __attribute__((ext_vector_type(4)))  unsigned int u32x4;
typedef __attribute__((ext_vector_type(8)))  int i32x8;
typedef __attribute__((ext_vector_type(4)))  int i32x4;

#define B_  4
#define S_  2048
#define D_  1024
#define H_  16
#define HD_ 64
#define NT_ (B_*S_)    // 8192 tokens
#define N3_ (3*D_)     // 3072
#define KP_ 256        // K-panel width staged in LDS per TDM load
#define NCH_ (D_/KP_)  // 4 panels
#define PANEL_ (64*KP_)

__device__ __forceinline__ v8f zero8() {
  v8f z = {0.f,0.f,0.f,0.f,0.f,0.f,0.f,0.f};
  return z;
}

__device__ __forceinline__ v8f wmma_bf16(v16bf a, v16bf b, v8f c) {
  return __builtin_amdgcn_wmma_f32_16x16x32_bf16(
      /*neg_a=*/false, a, /*neg_b=*/false, b,
      /*c_mod=*/(short)0, c, /*reuse_a=*/false, /*reuse_b=*/false);
}

// A-tile (16 rows x 32 K) from row-major bf16 src (stride elems/row).
// CDNA5 16-bit A layout: lanes 0-15 = rows, hold K {0..7,16..23};
// lanes 16-31 same rows, hold K {8..15,24..31}.
__device__ __forceinline__ v16bf load_a(const __bf16* src, int stride, int row0, int k0) {
  int lane = threadIdx.x & 31;
  int r = lane & 15, hf = lane >> 4;
  const __bf16* p = src + (size_t)(row0 + r) * stride + k0 + hf * 8;
  v8bf lo = *(const v8bf*)p;          // K = hf*8 + 0..7
  v8bf hi = *(const v8bf*)(p + 16);   // K = 16 + hf*8 + 0..7
  v16bf a;
#pragma unroll
  for (int i = 0; i < 8; ++i) { a[i] = lo[i]; a[i + 8] = hi[i]; }
  return a;
}

// B-tile (32 K x 16 cols) from column-major storage srcT[n][k] (strideK elems/col).
// CDNA5 16-bit B layout: lanes 0-15 = cols, hold K {0..7,16..23}; lanes 16-31 hold K {8..15,24..31}.
__device__ __forceinline__ v16bf load_b(const __bf16* srcT, int strideK, int n0, int k0) {
  int lane = threadIdx.x & 31;
  int r = lane & 15, hf = lane >> 4;
  const __bf16* p = srcT + (size_t)(n0 + r) * strideK + k0 + hf * 8;
  v8bf lo = *(const v8bf*)p;
  v8bf hi = *(const v8bf*)(p + 16);
  v16bf b;
#pragma unroll
  for (int i = 0; i < 8; ++i) { b[i] = lo[i]; b[i + 8] = hi[i]; }
  return b;
}

// ---------------- TDM: issue DMA of a 64(N) x 256(K) bf16 panel into LDS ----------------
// D# per CDNA5 ISA ch.8: group0 = {count/flags, lds_addr, global_addr, type=2},
// group1 = {mask/data_size, dims, tile dims, strides}. 2D tensor: groups 2/3/4 zero.
#if __has_builtin(__builtin_amdgcn_tensor_load_to_lds)
#define HAVE_TDM 1
__device__ __forceinline__ void tdm_issue(const __bf16* tile_base, int tensorN, __bf16* lds_dst) {
  u32x4 g0; i32x8 g1;
  i32x4 gz  = {0, 0, 0, 0};
  i32x8 gz8 = {0, 0, 0, 0, 0, 0, 0, 0};
  unsigned long long ga = (unsigned long long)tile_base;
  unsigned ldsoff = (unsigned)(unsigned long long)lds_dst;       // addr[31:0] = LDS offset
  g0[0] = 1u;                                          // count=1, user mode
  g0[1] = ldsoff;                                      // lds_addr
  g0[2] = (unsigned)(ga & 0xFFFFFFFFull);              // global_addr[31:0]
  g0[3] = (unsigned)((ga >> 32) & 0x01FFFFFFull) | (2u << 30);   // addr[56:32] | type=2
  g1[0] = 1 << 16;                                     // workgroup_mask=0, data_size=1 (2B)
  g1[1] = (D_ & 0xFFFF) << 16;                         // tensor_dim0 = 1024 (K)
  g1[2] = (tensorN & 0xFFFF) << 16;                    // dim0 hi=0 | tensor_dim1 lo (N)
  g1[3] = ((tensorN >> 16) & 0xFFFF) | (KP_ << 16);    // dim1 hi | tile_dim0 = 256
  g1[4] = 64;                                          // tile_dim1 = 64, tile_dim2 = 0
  g1[5] = D_;                                          // tensor_dim0_stride = 1024
  g1[6] = 0;
  g1[7] = 0;
  __builtin_amdgcn_tensor_load_to_lds(g0, g1, gz, gz, gz8, 0);
}
#else
#define HAVE_TDM 0
#endif

// ---------------- conversion kernels ----------------
__global__ void k_cvt(const float* __restrict__ in, __bf16* __restrict__ out, int n) {
  int i = blockIdx.x * 256 + threadIdx.x;
  if (i < n) out[i] = (__bf16)in[i];
}

// out[n*K + k] = in[k*N + n]   (weight transpose to col-major-for-B)
__global__ void k_cvt_t(const float* __restrict__ in, __bf16* __restrict__ out, int K, int N) {
  long long i = (long long)blockIdx.x * 256 + threadIdx.x;
  if (i < (long long)K * N) {
    int n = (int)(i / K), k = (int)(i % K);
    out[i] = (__bf16)in[(size_t)k * N + n];
  }
}

// 8 WMMAs per K-step: 32(M) x 64(N) per wave, B from LDS panel (stride KP_).
__device__ __forceinline__ void gemm_chunk(const __bf16* X, int row0, const __bf16* sB,
                                           int kglob0, v8f acc[8]) {
  for (int kb = 0; kb < KP_ / 32; ++kb) {
    v16bf a0 = load_a(X, D_, row0,      kglob0 + kb * 32);
    v16bf a1 = load_a(X, D_, row0 + 16, kglob0 + kb * 32);
#pragma unroll
    for (int j = 0; j < 4; ++j) {
      v16bf b = load_b(sB, KP_, j * 16, kb * 32);
      acc[j]     = wmma_bf16(a0, b, acc[j]);
      acc[4 + j] = wmma_bf16(a1, b, acc[4 + j]);
    }
  }
}

// Main loop: double-buffered TDM weight panels overlapped with WMMA compute.
__device__ __forceinline__ void gemm_main(const __bf16* __restrict__ X,
                                          const __bf16* __restrict__ Wt,
                                          int n0, int tensorN, int row0,
                                          __bf16* sB /* [2*PANEL_] */, v8f acc[8]) {
#if HAVE_TDM
  if (threadIdx.x == 0)
    tdm_issue(Wt + (size_t)n0 * D_, tensorN, sB);            // panel 0
  for (int ch = 0; ch < NCH_; ++ch) {
    __bf16* cur = sB + (ch & 1) * PANEL_;
    if (threadIdx.x == 0) {
      if (ch + 1 < NCH_) {
        tdm_issue(Wt + (size_t)n0 * D_ + (ch + 1) * KP_, tensorN,
                  sB + ((ch + 1) & 1) * PANEL_);             // prefetch next panel
        __builtin_amdgcn_s_wait_tensorcnt(1);                // panel ch landed (in-order)
      } else {
        __builtin_amdgcn_s_wait_tensorcnt(0);
      }
    }
    asm volatile("" ::: "memory");    // TDM wrote LDS behind the compiler's back
    __syncthreads();                  // panel ch visible to all waves
    gemm_chunk(X, row0, cur, ch * KP_, acc);
    __syncthreads();                  // all waves done before buffer 'cur' is reused
  }
#else
  for (int ch = 0; ch < NCH_; ++ch) {
    __syncthreads();
    for (int idx = threadIdx.x; idx < PANEL_ / 8; idx += 256) {
      int rown = idx >> 5, c8 = idx & 31;
      ((v8bf*)sB)[idx] = *(const v8bf*)(Wt + (size_t)(n0 + rown) * D_ + ch * KP_ + c8 * 8);
    }
    __syncthreads();
    gemm_chunk(X, row0, sB, ch * KP_, acc);
  }
#endif
}

// ---------------- QKV GEMM: [8192,1024] x [1024,3072] ----------------
// Block = 8 waves covering 256 M rows x 64 N cols; B panels staged by TDM into LDS.
__global__ __launch_bounds__(256)
void k_gemm_qkv(const __bf16* __restrict__ X, const __bf16* __restrict__ Wt,
                const float* __restrict__ bias,
                __bf16* __restrict__ Q, __bf16* __restrict__ K, __bf16* __restrict__ Vt) {
  __shared__ __bf16 sB[2 * PANEL_];          // 64 KB, double buffered
  int nt = blockIdx.y;                        // 0..47 (64-wide N tiles)
  int row0 = blockIdx.x * 256 + (threadIdx.x >> 5) * 32;
  int lane = threadIdx.x & 31;
  int r = lane & 15, hf = lane >> 4;

  v8f acc[8];
#pragma unroll
  for (int j = 0; j < 8; ++j) acc[j] = zero8();

  gemm_main(X, Wt, nt * 64, N3_, row0, sB, acc);

#pragma unroll
  for (int j = 0; j < 4; ++j) {
    int n = nt * 64 + j * 16 + r;             // C layout: col = lane%16
    float bv = bias[n];
#pragma unroll
    for (int half = 0; half < 2; ++half) {
#pragma unroll
      for (int i = 0; i < 8; ++i) {
        int m = row0 + half * 16 + i + 8 * hf;
        float val = acc[half * 4 + j][i] + bv;
        __bf16 o = (__bf16)val;
        int b = m >> 11, s = m & (S_ - 1);
        int nn = n;
        if (nn < D_) {                        // Q : [b,h,s,hd]
          int hh = nn >> 6, hd = nn & 63;
          Q[(((size_t)(b * H_ + hh)) * S_ + s) * HD_ + hd] = o;
        } else if (nn < 2 * D_) {             // K : [b,h,s,hd] (B-layout for QK^T)
          nn -= D_;
          int hh = nn >> 6, hd = nn & 63;
          K[(((size_t)(b * H_ + hh)) * S_ + s) * HD_ + hd] = o;
        } else {                              // V transposed: [b,h,hd,s] (B-layout for PV)
          nn -= 2 * D_;
          int hh = nn >> 6, hd = nn & 63;
          Vt[(((size_t)(b * H_ + hh)) * HD_ + hd) * S_ + s] = o;
        }
      }
    }
  }
}

// ---------------- output projection: [8192,1024] x [1024,1024] + bias (fp32 out) ----------------
__global__ __launch_bounds__(256)
void k_gemm_proj(const __bf16* __restrict__ A, const __bf16* __restrict__ Wt,
                 const float* __restrict__ bias, float* __restrict__ out) {
  __shared__ __bf16 sB[2 * PANEL_];          // 64 KB, double buffered
  int nt = blockIdx.y;                        // 0..15
  int row0 = blockIdx.x * 256 + (threadIdx.x >> 5) * 32;
  int lane = threadIdx.x & 31;
  int r = lane & 15, hf = lane >> 4;

  v8f acc[8];
#pragma unroll
  for (int j = 0; j < 8; ++j) acc[j] = zero8();

  gemm_main(A, Wt, nt * 64, D_, row0, sB, acc);

#pragma unroll
  for (int j = 0; j < 4; ++j) {
    int n = nt * 64 + j * 16 + r;
    float bv = bias[n];
#pragma unroll
    for (int half = 0; half < 2; ++half) {
#pragma unroll
      for (int i = 0; i < 8; ++i) {
        int m = row0 + half * 16 + i + 8 * hf;
        out[(size_t)m * D_ + n] = acc[half * 4 + j][i] + bv;
      }
    }
  }
}

// ---------------- flash attention: one 16-row query tile per wave ----------------
__global__ __launch_bounds__(32)
void k_attn(const __bf16* __restrict__ Q, const __bf16* __restrict__ K,
            const __bf16* __restrict__ Vt, __bf16* __restrict__ ctx) {
  __shared__ __bf16 P[16 * 32];              // P tile, C-layout -> A-layout staging
  int qt = blockIdx.x;                        // 0..127 query tiles
  int bh = blockIdx.y;                        // 0..63  (b,h)
  int b = bh >> 4, h = bh & 15;
  const __bf16* qh = Q + (size_t)bh * S_ * HD_;
  const __bf16* kh = K + (size_t)bh * S_ * HD_;
  const __bf16* vh = Vt + (size_t)bh * HD_ * S_;
  int lane = threadIdx.x & 31;
  int r = lane & 15, hf = lane >> 4;

  // Q tile (16x64), pre-scaled by 1/sqrt(64) (exact in bf16)
  v16bf qa0 = load_a(qh, HD_, qt * 16, 0);
  v16bf qa1 = load_a(qh, HD_, qt * 16, 32);
#pragma unroll
  for (int i = 0; i < 16; ++i) {
    qa0[i] = (__bf16)((float)qa0[i] * 0.125f);
    qa1[i] = (__bf16)((float)qa1[i] * 0.125f);
  }

  float Mi[8], Li[8];
#pragma unroll
  for (int i = 0; i < 8; ++i) { Mi[i] = -__builtin_inff(); Li[i] = 0.f; }
  v8f O[4];
#pragma unroll
  for (int j = 0; j < 4; ++j) O[j] = zero8();

  int nkb = qt / 2 + 1;                       // causal: keys 0 .. qt*16+15
  for (int kb = 0; kb < nkb; ++kb) {
    int key0 = kb * 32;

    // scores: 16 queries x 32 keys, reduce over hd=64 -> 4 WMMAs
    v8f s0 = zero8(), s1 = zero8();
    s0 = wmma_bf16(qa0, load_b(kh, HD_, key0, 0), s0);
    s0 = wmma_bf16(qa1, load_b(kh, HD_, key0, 32), s0);
    s1 = wmma_bf16(qa0, load_b(kh, HD_, key0 + 16, 0), s1);
    s1 = wmma_bf16(qa1, load_b(kh, HD_, key0 + 16, 32), s1);

    __syncthreads();                          // P reads of previous iter complete
#pragma unroll
    for (int i = 0; i < 8; ++i) {
      int m = i + 8 * hf;
      int qrow = qt * 16 + m;
      float e0 = (key0 + r      <= qrow) ? s0[i] : -__builtin_inff();
      float e1 = (key0 + 16 + r <= qrow) ? s1[i] : -__builtin_inff();
      float mx = fmaxf(e0, e1);
#pragma unroll
      for (int off = 1; off < 16; off <<= 1)  // reduce within 16-lane half
        mx = fmaxf(mx, __shfl_xor(mx, off, 32));
      float Mn = fmaxf(Mi[i], mx);
      float alpha = __expf(Mi[i] - Mn);
      float p0 = __expf(e0 - Mn);
      float p1 = __expf(e1 - Mn);
      float rs = p0 + p1;
#pragma unroll
      for (int off = 1; off < 16; off <<= 1)
        rs += __shfl_xor(rs, off, 32);
      Li[i] = Li[i] * alpha + rs;
      Mi[i] = Mn;
#pragma unroll
      for (int j = 0; j < 4; ++j) O[j][i] *= alpha;
      P[m * 32 + r]      = (__bf16)p0;        // row-major P tile in LDS
      P[m * 32 + 16 + r] = (__bf16)p1;
    }
    __syncthreads();                          // stores visible before A-layout reload

    v16bf pa = load_a(&P[0], 32, 0, 0);       // P in A-layout (16x32)
#pragma unroll
    for (int j = 0; j < 4; ++j)               // O(16x64) += P(16x32) x V(32x64)
      O[j] = wmma_bf16(pa, load_b(vh, S_, j * 16, key0), O[j]);
  }

  size_t tok = (size_t)b * S_ + qt * 16;
#pragma unroll
  for (int i = 0; i < 8; ++i) {
    int m = i + 8 * hf;
    float inv = 1.0f / Li[i];
    __bf16* dst = ctx + (tok + m) * D_ + h * HD_;
#pragma unroll
    for (int j = 0; j < 4; ++j)
      dst[j * 16 + r] = (__bf16)(O[j][i] * inv);
  }
}

extern "C" void kernel_launch(void* const* d_in, const int* in_sizes, int n_in,
                              void* d_out, int out_size, void* d_ws, size_t ws_size,
                              hipStream_t stream) {
  const float* hs    = (const float*)d_in[0];  // [4,2048,1024]
  const float* wqkv  = (const float*)d_in[1];  // [1024,3072]
  const float* bqkv  = (const float*)d_in[2];  // [3072]
  const float* wproj = (const float*)d_in[3];  // [1024,1024]
  const float* bproj = (const float*)d_in[4];  // [1024]
  float* out = (float*)d_out;                  // [4,2048,1024]

  __bf16* X16 = (__bf16*)d_ws;                          // 8192*1024
  __bf16* WQt = X16 + (size_t)NT_ * D_;                 // 3072*1024 (transposed)
  __bf16* WPt = WQt + (size_t)N3_ * D_;                 // 1024*1024 (transposed)
  __bf16* Qb  = WPt + (size_t)D_ * D_;                  // [B,H,S,64]
  __bf16* Kb  = Qb  + (size_t)NT_ * D_;                 // [B,H,S,64]
  __bf16* Vt  = Kb  + (size_t)NT_ * D_;                 // [B,H,64,S]
  __bf16* Ctx = Vt  + (size_t)NT_ * D_;                 // [B,S,D]

  k_cvt  <<<(NT_ * D_) / 256, 256, 0, stream>>>(hs, X16, NT_ * D_);
  k_cvt_t<<<(N3_ * D_) / 256, 256, 0, stream>>>(wqkv, WQt, D_, N3_);
  k_cvt_t<<<(D_ * D_)  / 256, 256, 0, stream>>>(wproj, WPt, D_, D_);

  k_gemm_qkv <<<dim3(NT_ / 256, N3_ / 64), 256, 0, stream>>>(X16, WQt, bqkv, Qb, Kb, Vt);
  k_attn     <<<dim3(S_ / 16, B_ * H_),    32,  0, stream>>>(Qb, Kb, Vt, Ctx);
  k_gemm_proj<<<dim3(NT_ / 256, D_ / 64),  256, 0, stream>>>(Ctx, WPt, bproj, out);
}
